// PCTNet_17300128268588
// MI455X (gfx1250) — matrix operand order repaired and checked by reading, not compiled, not run on hardware
//
#include <hip/hip_runtime.h>
#include <hip/hip_bf16.h>

typedef __attribute__((ext_vector_type(16))) _Float16 v16h;
typedef __attribute__((ext_vector_type(8)))  _Float16 v8h;
typedef __attribute__((ext_vector_type(8)))  float    v8f;

#define BATCH 8
#define CCH   256
#define NPB   1024          // tokens per batch (32*32)
#define NPIX  (BATCH*NPB)   // 8192
#define KCONV 2304          // 256*9
#define HEADS 8
#define DH    32
#define NBIAS 3969          // 63*63

__device__ __forceinline__ v8f wmma16(v16h a, v16h b, v8f c) {
  return __builtin_amdgcn_wmma_f32_16x16x32_f16(false, a, false, b, (short)0, c,
                                                false, false);
}

// A fragment: 16x32 f16 from row-major [m][k] tile (ld halves).
// Lane layout (ISA 7.12.2): lane half hi holds k = {0..7,16..23} + 8*hi for row m=lane&15.
// Two contiguous 8-half runs -> two ds_load_b128.
__device__ __forceinline__ v16h frag_a(const _Float16* src, int ld) {
  const int lane = threadIdx.x & 31;
  const int m   = lane & 15;
  const int off = (lane >> 4) << 3;
  const v8h lo = *(const v8h*)(src + m * ld + off);
  const v8h hi = *(const v8h*)(src + m * ld + 16 + off);
  return __builtin_shufflevector(lo, hi, 0, 1, 2, 3, 4, 5, 6, 7,
                                 8, 9, 10, 11, 12, 13, 14, 15);
}

// B fragment: 32x16 f16 where the tile is stored N-major: B[k][n] = src[n*ld + k].
// Lane needs k = {0..7,16..23} + 8*hi for column n=lane&15 -> two v8h loads.
__device__ __forceinline__ v16h frag_bt(const _Float16* src, int ld) {
  const int lane = threadIdx.x & 31;
  const int n  = lane & 15;
  const int kb = (lane >> 4) << 3;
  const v8h lo = *(const v8h*)(src + n * ld + kb);
  const v8h hi = *(const v8h*)(src + n * ld + 16 + kb);
  return __builtin_shufflevector(lo, hi, 0, 1, 2, 3, 4, 5, 6, 7,
                                 8, 9, 10, 11, 12, 13, 14, 15);
}

// 16 consecutive f32 -> v16h (float4 global loads + cvt).
__device__ __forceinline__ v16h cvt16(const float* __restrict__ src) {
  const float4* s4 = (const float4*)src;
  const float4 a0 = s4[0], a1 = s4[1], a2 = s4[2], a3 = s4[3];
  v16h r;
  r[0] = (_Float16)a0.x;  r[1] = (_Float16)a0.y;
  r[2] = (_Float16)a0.z;  r[3] = (_Float16)a0.w;
  r[4] = (_Float16)a1.x;  r[5] = (_Float16)a1.y;
  r[6] = (_Float16)a1.z;  r[7] = (_Float16)a1.w;
  r[8] = (_Float16)a2.x;  r[9] = (_Float16)a2.y;
  r[10] = (_Float16)a2.z; r[11] = (_Float16)a2.w;
  r[12] = (_Float16)a3.x; r[13] = (_Float16)a3.y;
  r[14] = (_Float16)a3.z; r[15] = (_Float16)a3.w;
  return r;
}

__global__ void pct_zero(float* p, int n) {
  int i = blockIdx.x * blockDim.x + threadIdx.x;
  if (i < n) p[i] = 0.f;
}

// bias_table [NBIAS][8] -> head-major [8][NBIAS]
__global__ void pct_bias_repack(const float* __restrict__ bt, float* __restrict__ br) {
  int i = blockIdx.x * blockDim.x + threadIdx.x;
  if (i < NBIAS * HEADS) {
    const int idx = i >> 3, h = i & 7;
    br[h * NBIAS + idx] = bt[idx * 8 + h];
  }
}

// Implicit-GEMM 3x3 conv: M=256 x N=8192 x K=2304.
// grid (NPIX/64, CCH/64, 3), block 128 (4 waves).
__global__ void pct_conv(const float* __restrict__ x,
                         const float* __restrict__ wq, const float* __restrict__ wk,
                         const float* __restrict__ wv,
                         _Float16* __restrict__ yq, _Float16* __restrict__ yk,
                         _Float16* __restrict__ yv,
                         float* __restrict__ stats /*[3][B][2]*/) {
  const int which = blockIdx.z;
  const float* wsel = which == 0 ? wq : (which == 1 ? wk : wv);
  _Float16*    ysel = which == 0 ? yq : (which == 1 ? yk : yv);

  __shared__ __align__(32) _Float16 As[64 * 32];   // [m][k]
  __shared__ __align__(32) _Float16 Bt[64 * 32];   // [n][k] (transposed)
  __shared__ float redS[4], redQ[4];

  const int tid  = threadIdx.x;
  const int lane = tid & 31, wave = tid >> 5;
  const int tn = blockIdx.x * 64;
  const int tm = blockIdx.y * 64;
  const int b  = tn >> 10;

  v8f acc[4] = {};

  const int ar  = tid >> 1;            // A fill: row
  const int as  = (tid & 1) << 4;      // A fill: k segment

  for (int k0 = 0; k0 < KCONV; k0 += 32) {
    if (k0 + 32 < KCONV)
      __builtin_prefetch(wsel + (size_t)(tm + ar) * KCONV + k0 + 32, 0, 0);
    // A tile: 16 contiguous f32 per thread -> one v16h LDS store
    *(v16h*)(&As[ar * 32 + as]) =
        cvt16(wsel + (size_t)(tm + ar) * KCONV + k0 + as);
    // B tile (im2col gather, boundary-checked) stored N-major
    for (int i = tid; i < 32 * 64; i += 128) {
      const int kk = i >> 6, nn = i & 63;
      const int kf = k0 + kk;
      const int ci = kf / 9, rem = kf % 9;
      const int kh = rem / 3, kw = rem % 3;
      const int hw = (tn + nn) & 1023;
      const int hh = (hw >> 5) + kh - 1, ww = (hw & 31) + kw - 1;
      float v = 0.f;
      if ((unsigned)hh < 32u && (unsigned)ww < 32u)
        v = x[(((size_t)b * CCH + ci) << 10) + (hh << 5) + ww];
      Bt[nn * 32 + kk] = (_Float16)v;
    }
    __syncthreads();
    const v16h a = frag_a(&As[wave * 16 * 32], 32);
#pragma unroll
    for (int f = 0; f < 4; ++f) {
      const v16h bf = frag_bt(&Bt[(f * 16) * 32], 32);
      acc[f] = wmma16(a, bf, acc[f]);
    }
    __syncthreads();
  }

  // epilogue: f16 store [b][co][hw] + GroupNorm partial stats
  const int hi = lane >> 4, nl = lane & 15;
  float ps = 0.f, pq = 0.f;
#pragma unroll
  for (int f = 0; f < 4; ++f) {
    const int hw = (tn + f * 16 + nl) & 1023;
#pragma unroll
    for (int r = 0; r < 8; ++r) {
      const int co = tm + wave * 16 + hi * 8 + r;
      const float v = acc[f][r];
      ysel[(((size_t)b * CCH + co) << 10) + hw] = (_Float16)v;
      ps += v; pq += v * v;
    }
  }
  for (int off = 16; off; off >>= 1) {
    ps += __shfl_down(ps, off);
    pq += __shfl_down(pq, off);
  }
  if (lane == 0) { redS[wave] = ps; redQ[wave] = pq; }
  __syncthreads();
  if (tid == 0) {
    atomicAdd(&stats[(which * BATCH + b) * 2 + 0], redS[0] + redS[1] + redS[2] + redS[3]);
    atomicAdd(&stats[(which * BATCH + b) * 2 + 1], redQ[0] + redQ[1] + redQ[2] + redQ[3]);
  }
}

// GroupNorm(1) + GELU(exact) + repack. q,k -> [bh][n][d]; v -> [bh][d][n] (d-major
// so the attention PV B-tile fill is a straight vector copy).
__global__ void pct_gnpack(const _Float16* __restrict__ yq, const _Float16* __restrict__ yk,
                           const _Float16* __restrict__ yv,
                           const float* gq, const float* bq, const float* gk,
                           const float* bk, const float* gv, const float* bv,
                           const float* __restrict__ stats,
                           _Float16* __restrict__ qh, _Float16* __restrict__ kh,
                           _Float16* __restrict__ vh) {
  const int idx = blockIdx.x * blockDim.x + threadIdx.x;   // 3 * 2^21 total
  const int which = idx >> 21;
  const int r  = idx & ((1 << 21) - 1);
  const int b  = r >> 18;
  const int cz = (r >> 10) & 255;
  const int hw = r & 1023;

  const _Float16* y  = which == 0 ? yq : (which == 1 ? yk : yv);
  const float*    g  = which == 0 ? gq : (which == 1 ? gk : gv);
  const float*    bb = which == 0 ? bq : (which == 1 ? bk : bv);

  const float cnt  = 1.f / (float)(CCH * NPB);
  const float mean = stats[(which * BATCH + b) * 2 + 0] * cnt;
  const float var  = stats[(which * BATCH + b) * 2 + 1] * cnt - mean * mean;
  const float inv  = rsqrtf(var + 1e-6f);

  const float xn = ((float)y[r] - mean) * inv * g[cz] + bb[cz];
  const float ge = 0.5f * xn * (1.f + erff(xn * 0.70710678118f));

  const int head = cz >> 5, dd = cz & 31;
  const size_t bh = (size_t)b * HEADS + head;
  if (which == 2) {
    vh[((bh << 5) + dd) * NPB + hw] = (_Float16)ge;              // [bh][d][n]
  } else {
    _Float16* d = which == 0 ? qh : kh;
    d[((bh << 10) + hw) * DH + dd] = (_Float16)ge;               // [bh][n][d]
  }
}

// Flash attention: grid (1024/64, B*HEADS), block 128 (4 waves, 16 rows each).
__global__ void pct_attn(const _Float16* __restrict__ qh, const _Float16* __restrict__ kh,
                         const _Float16* __restrict__ vh,
                         const float* __restrict__ biasr /* [8][NBIAS] */,
                         _Float16* __restrict__ oh) {
  const int bh   = blockIdx.y;
  const int head = bh & (HEADS - 1);
  const int row0 = blockIdx.x * 64;
  const int tid  = threadIdx.x, lane = tid & 31, wave = tid >> 5;
  const int hi = lane >> 4, nl = lane & 15;

  __shared__ __align__(32) _Float16 Qs[64 * DH];     // [n][d]
  __shared__ __align__(32) _Float16 Ks[64 * DH];     // [key][d]  (N-major for QK^T)
  __shared__ __align__(32) _Float16 Vt[DH * 64];     // [d][key]  (N-major for PV)
  __shared__ __align__(32) _Float16 Ps[4][16 * 64];  // per-wave P [m][j]

  const _Float16* qb = qh + ((size_t)bh << 10) * DH;
  const _Float16* kb = kh + ((size_t)bh << 10) * DH;
  const _Float16* vb = vh + ((size_t)bh << 5) * NPB;
  const float*    bt = biasr + (size_t)head * NBIAS;

  ((v16h*)Qs)[tid] = ((const v16h*)(qb + (size_t)row0 * DH))[tid];
  __syncthreads();

  const v16h aq = frag_a(&Qs[wave * 16 * DH], DH);   // 16 Q rows, K=d=32
  v8f o0 = {}, o1 = {};
  float rm[8], rl[8];
#pragma unroll
  for (int r = 0; r < 8; ++r) { rm[r] = -1e30f; rl[r] = 0.f; }
  const v8f zero = {};

  for (int c0 = 0; c0 < NPB; c0 += 64) {
    // K tile: straight 32B vector copy; V tile: d-major rows, vector copy
    ((v16h*)Ks)[tid] = ((const v16h*)(kb + (size_t)c0 * DH))[tid];
    {
      const int dd = tid >> 2, seg = (tid & 3) << 4;
      *(v16h*)(&Vt[dd * 64 + seg]) =
          *(const v16h*)(vb + (size_t)dd * NPB + c0 + seg);
    }
    __syncthreads();

    // S = Q K^T : per-wave 16x64, 4 col fragments (B[k][n] = Ks[n*32 + k])
    v8f s[4];
#pragma unroll
    for (int f = 0; f < 4; ++f) {
      const v16h bkf = frag_bt(&Ks[(f * 16) * DH], DH);
      s[f] = wmma16(aq, bkf, zero);
    }

    // relative position bias (head-major table)
#pragma unroll
    for (int f = 0; f < 4; ++f) {
      const int j = c0 + f * 16 + nl;
      const int yj = j >> 5, xj = j & 31;
#pragma unroll
      for (int r = 0; r < 8; ++r) {
        const int i  = row0 + wave * 16 + hi * 8 + r;
        const int yi = i >> 5, xi = i & 31;
        s[f][r] += bt[(yi - yj + 31) * 63 + (xi - xj + 31)];
      }
    }

    // online softmax (row = (r,hi); 16 lanes of each half share a row)
#pragma unroll
    for (int r = 0; r < 8; ++r) {
      float mx = fmaxf(fmaxf(s[0][r], s[1][r]), fmaxf(s[2][r], s[3][r]));
#pragma unroll
      for (int off = 8; off; off >>= 1) mx = fmaxf(mx, __shfl_xor(mx, off, 16));
      const float nm = fmaxf(rm[r], mx);
      const float corr = __expf(rm[r] - nm);
      rm[r] = nm;
      float psum = 0.f;
#pragma unroll
      for (int f = 0; f < 4; ++f) {
        const float p = __expf(s[f][r] - nm);
        s[f][r] = p;
        psum += p;
      }
#pragma unroll
      for (int off = 8; off; off >>= 1) psum += __shfl_xor(psum, off, 16);
      rl[r] = rl[r] * corr + psum;
      o0[r] *= corr;
      o1[r] *= corr;
    }

    // P -> per-wave LDS buffer (f16)
#pragma unroll
    for (int f = 0; f < 4; ++f)
#pragma unroll
      for (int r = 0; r < 8; ++r)
        Ps[wave][(r + hi * 8) * 64 + f * 16 + nl] = (_Float16)s[f][r];
    __syncthreads();

    // O += P V : K=64 (two chunks), N=32 (two col fragments); B[k][n] = Vt[n*64 + k]
#pragma unroll
    for (int kc = 0; kc < 2; ++kc) {
      const v16h ap = frag_a(&Ps[wave][kc * 32], 64);
      const v16h b0 = frag_bt(&Vt[0 * 64 + kc * 32], 64);
      const v16h b1 = frag_bt(&Vt[16 * 64 + kc * 32], 64);
      o0 = wmma16(ap, b0, o0);
      o1 = wmma16(ap, b1, o1);
    }
    __syncthreads();
  }

  // normalize and store [bh][n][d]
#pragma unroll
  for (int r = 0; r < 8; ++r) {
    const int i = row0 + wave * 16 + hi * 8 + r;
    const float invl = 1.f / rl[r];
    oh[(((size_t)bh << 10) + i) * DH + nl]      = (_Float16)(o0[r] * invl);
    oh[(((size_t)bh << 10) + i) * DH + 16 + nl] = (_Float16)(o1[r] * invl);
  }
}

// Output projection: M=256 x N=8192 x K=256 + bias, f32 out.
__global__ void pct_proj(const float* __restrict__ wo, const float* __restrict__ bo,
                         const _Float16* __restrict__ oh, float* __restrict__ out) {
  __shared__ __align__(32) _Float16 As[64 * 32];   // [m][k]
  __shared__ __align__(32) _Float16 Bt[64 * 32];   // [n][k]

  const int tid  = threadIdx.x;
  const int lane = tid & 31, wave = tid >> 5;
  const int tn = blockIdx.x * 64;
  const int tm = blockIdx.y * 64;
  const int b  = tn >> 10;

  v8f acc[4] = {};

  const int ar = tid >> 1, as = (tid & 1) << 4;

  for (int k0 = 0; k0 < CCH; k0 += 32) {
    // A: weights, 16 contiguous f32 -> v16h
    *(v16h*)(&As[ar * 32 + as]) = cvt16(wo + (size_t)(tm + ar) * CCH + k0 + as);
    // B: attention output; k-chunk of 32 spans exactly one head -> contiguous d run
    {
      const int nn = tid >> 1, seg = (tid & 1) << 4;
      const int head = k0 >> 5;
      const int hw = (tn + nn) & 1023;
      *(v16h*)(&Bt[nn * 32 + seg]) =
          *(const v16h*)(oh + ((((size_t)b * HEADS + head) << 10) + hw) * DH + seg);
    }
    __syncthreads();
    const v16h a = frag_a(&As[wave * 16 * 32], 32);
#pragma unroll
    for (int f = 0; f < 4; ++f) {
      const v16h bf = frag_bt(&Bt[(f * 16) * 32], 32);
      acc[f] = wmma16(a, bf, acc[f]);
    }
    __syncthreads();
  }

  const int hi = lane >> 4, nl = lane & 15;
#pragma unroll
  for (int f = 0; f < 4; ++f) {
    const int hw = (tn + f * 16 + nl) & 1023;
#pragma unroll
    for (int r = 0; r < 8; ++r) {
      const int co = tm + wave * 16 + hi * 8 + r;
      out[(((size_t)b * CCH + co) << 10) + hw] = acc[f][r] + bo[co];
    }
  }
}

extern "C" void kernel_launch(void* const* d_in, const int* in_sizes, int n_in,
                              void* d_out, int out_size, void* d_ws, size_t ws_size,
                              hipStream_t stream) {
  const float* x   = (const float*)d_in[0];
  const float* wq  = (const float*)d_in[1];
  const float* wk  = (const float*)d_in[2];
  const float* wv  = (const float*)d_in[3];
  const float* gq  = (const float*)d_in[4];
  const float* bq  = (const float*)d_in[5];
  const float* gk  = (const float*)d_in[6];
  const float* bk  = (const float*)d_in[7];
  const float* gv  = (const float*)d_in[8];
  const float* bv  = (const float*)d_in[9];
  const float* bias_table = (const float*)d_in[10];
  const float* wo  = (const float*)d_in[11];
  const float* bo  = (const float*)d_in[12];
  float* out = (float*)d_out;

  char* ws = (char*)d_ws;
  const size_t hb = (size_t)2097152 * sizeof(_Float16);  // 4 MB per f16 buffer
  _Float16* yq = (_Float16*)(ws + 0 * hb);
  _Float16* yk = (_Float16*)(ws + 1 * hb);
  _Float16* yv = (_Float16*)(ws + 2 * hb);
  _Float16* qh = (_Float16*)(ws + 3 * hb);
  _Float16* kh = (_Float16*)(ws + 4 * hb);
  _Float16* vh = (_Float16*)(ws + 5 * hb);
  _Float16* oh = (_Float16*)(ws + 6 * hb);
  float* stats = (float*)(ws + 7 * hb);              // 48 floats
  float* biasr = (float*)(ws + 7 * hb + 256);        // [8][NBIAS]

  pct_zero<<<1, 64, 0, stream>>>(stats, 48);
  pct_bias_repack<<<(NBIAS * HEADS + 255) / 256, 256, 0, stream>>>(bias_table, biasr);

  dim3 gconv(NPIX / 64, CCH / 64, 3);
  pct_conv<<<gconv, 128, 0, stream>>>(x, wq, wk, wv, yq, yk, yv, stats);

  pct_gnpack<<<(3 * BATCH * CCH * NPB) / 256, 256, 0, stream>>>(
      yq, yk, yv, gq, bq, gk, bk, gv, bv, stats, qh, kh, vh);

  dim3 gattn(NPB / 64, BATCH * HEADS);
  pct_attn<<<gattn, 128, 0, stream>>>(qh, kh, vh, biasr, oh);

  dim3 gproj(NPIX / 64, CCH / 64);
  pct_proj<<<gproj, 128, 0, stream>>>(wo, bo, oh, out);
}
